// PyramidMSDeformAttnTransformerEncoderOnly_87419764343060
// MI455X (gfx1250) — matrix operand order repaired and checked by reading, not compiled, not run on hardware
//
#include <hip/hip_runtime.h>
#include <math.h>

// ---------------------------------------------------------------------------
// Types for WMMA fragments (wave32, gfx1250)
// ---------------------------------------------------------------------------
typedef __attribute__((ext_vector_type(16))) _Float16 v16h;
typedef __attribute__((ext_vector_type(8)))  _Float16 h8;
typedef __attribute__((ext_vector_type(8)))  float    v8f;

// ---------------------------------------------------------------------------
// Problem constants
// ---------------------------------------------------------------------------
#define L5 576
#define L4 2304
#define L3 9216
#define LT 12096          // 576+2304+9216
#define DM 256
#define NB 2              // batch
#define NH 8              // heads
#define NP 4              // points

__device__ __forceinline__ float waveReduceSum(float v) {
    for (int m = 16; m > 0; m >>= 1) v += __shfl_xor(v, m, 32);
    return v;
}

// ---------------------------------------------------------------------------
// Weight repack: W (K,N) f32 row-major -> WT (N,K) f16 row-major
// ---------------------------------------------------------------------------
__global__ void k_pack_weight(const float* __restrict__ W, _Float16* __restrict__ WT,
                              int K, int N) {
    size_t i = (size_t)blockIdx.x * 256 + threadIdx.x;
    size_t tot = (size_t)K * N;
    if (i >= tot) return;
    int k = (int)(i % K);
    int n = (int)(i / K);
    WT[i] = (_Float16)W[(size_t)k * N + n];
}

// ---------------------------------------------------------------------------
// bulk f32 -> f16 convert
// ---------------------------------------------------------------------------
__global__ void k_cvt_f16(const float* __restrict__ in, _Float16* __restrict__ out,
                          size_t n) {
    size_t i = (size_t)blockIdx.x * 256 + threadIdx.x;
    if (i < n) out[i] = (_Float16)in[i];
}

// ---------------------------------------------------------------------------
// flatten: (B,256,L) -> (B,L,256) token-major
// ---------------------------------------------------------------------------
__global__ void k_flatten(const float* __restrict__ in, float* __restrict__ out, int L) {
    size_t i = (size_t)blockIdx.x * 256 + threadIdx.x;
    size_t tot = (size_t)NB * L * DM;
    if (i >= tot) return;
    int c = (int)(i % DM);
    size_t t = i / DM;
    int l = (int)(t % L);
    int b = (int)(t / L);
    out[i] = in[((size_t)b * DM + c) * L + l];
}

// flatten + level embed, writing into the full (B, LT, 256) pos buffer region
__global__ void k_flatten_pos(const float* __restrict__ in, const float* __restrict__ emb,
                              float* __restrict__ out_full, int L, int start) {
    size_t i = (size_t)blockIdx.x * 256 + threadIdx.x;
    size_t tot = (size_t)NB * L * DM;
    if (i >= tot) return;
    int c = (int)(i % DM);
    size_t t = i / DM;
    int l = (int)(t % L);
    int b = (int)(t / L);
    out_full[((size_t)b * LT + start + l) * DM + c] =
        in[((size_t)b * DM + c) * L + l] + emb[c];
}

// ---------------------------------------------------------------------------
// TRC resize: res3 (B,9216,256) raw-viewed as (B,256,96,96), bilinear
// (align_corners=False, edge clamp) -> out channel-major (B,256,OH,OH)
// ---------------------------------------------------------------------------
__global__ void k_resize(const float* __restrict__ prev, float* __restrict__ out, int OH) {
    int OW = OH;
    size_t i = (size_t)blockIdx.x * 256 + threadIdx.x;
    size_t tot = (size_t)NB * DM * OH * OW;
    if (i >= tot) return;
    int ox = (int)(i % OW);
    size_t t = i / OW;
    int oy = (int)(t % OH); t /= OH;
    int ch = (int)(t % DM);
    int b  = (int)(t / DM);
    float s = 96.0f / (float)OH;
    float sx = (ox + 0.5f) * s - 0.5f;
    float sy = (oy + 0.5f) * s - 0.5f;
    float fx = floorf(sx), fy = floorf(sy);
    int x0 = (int)fx, y0 = (int)fy;
    float wx = sx - fx, wy = sy - fy;
    int x0c = min(max(x0, 0), 95),   x1c = min(max(x0 + 1, 0), 95);
    int y0c = min(max(y0, 0), 95),   y1c = min(max(y0 + 1, 0), 95);
    const float* pb = prev + (size_t)b * L3 * DM + (size_t)ch * L3;
    float v00 = pb[y0c * 96 + x0c], v10 = pb[y0c * 96 + x1c];
    float v01 = pb[y1c * 96 + x0c], v11 = pb[y1c * 96 + x1c];
    out[i] = (1.f - wx) * (1.f - wy) * v00 + wx * (1.f - wy) * v10 +
             (1.f - wx) * wy * v01 + wx * wy * v11;
}

// ---------------------------------------------------------------------------
// TRC gate: gate[b,l] = sigmoid( dot(r[b, l*256 : l*256+256], Wt) + bt )
// (the raw .view() reinterpretation makes r channel-major flat), then
// out[b,l,c] = curr[b,l,c] * gate.  One wave per token.
// ---------------------------------------------------------------------------
__global__ __launch_bounds__(256)
void k_trc_gate(const float* __restrict__ curr, const float* __restrict__ r,
                const float* __restrict__ Wt, const float* __restrict__ bt,
                float* __restrict__ out, int L) {
    int lane = threadIdx.x & 31;
    int wv   = threadIdx.x >> 5;
    int t = blockIdx.x * 8 + wv;
    if (t >= NB * L) return;
    int b = t / L, l = t % L;
    const float* rr = r + (size_t)b * DM * L + (size_t)l * DM;
    float s = 0.f;
    #pragma unroll
    for (int i = 0; i < 8; ++i) {
        int c = lane + i * 32;
        s += rr[c] * Wt[c];
    }
    s = waveReduceSum(s);
    float gate = 1.f / (1.f + expf(-(s + bt[0])));
    const float* cu = curr + (size_t)t * DM;
    float* ou = out + (size_t)t * DM;
    #pragma unroll
    for (int i = 0; i < 8; ++i) {
        int c = lane + i * 32;
        ou[c] = cu[c] * gate;
    }
}

// ---------------------------------------------------------------------------
// q16 = (f16)(x + pos)  (q only ever feeds WMMA GEMMs)
// ---------------------------------------------------------------------------
__global__ void k_add_pos(const float* __restrict__ x, const float* __restrict__ posf,
                          _Float16* __restrict__ q, int Lcur) {
    size_t i = (size_t)blockIdx.x * 256 + threadIdx.x;
    size_t tot = (size_t)NB * Lcur * DM;
    if (i >= tot) return;
    int c = (int)(i % DM);
    size_t t = i / DM;
    int l = (int)(t % Lcur);
    int b = (int)(t / Lcur);
    q[i] = (_Float16)(x[i] + posf[((size_t)b * LT + l) * DM + c]);
}

// ---------------------------------------------------------------------------
// copy rows: dst[b*dstL + dstOff + l] = src[b*srcL + l], rows of 256 f32
// ---------------------------------------------------------------------------
__global__ void k_copy_rows(const float* __restrict__ src, float* __restrict__ dst,
                            int n, int srcL, int dstL, int dstOff) {
    size_t i = (size_t)blockIdx.x * 256 + threadIdx.x;
    size_t tot = (size_t)NB * n * DM;
    if (i >= tot) return;
    int c = (int)(i % DM);
    size_t t = i / DM;
    int l = (int)(t % n);
    int b = (int)(t / n);
    dst[((size_t)b * dstL + dstOff + l) * DM + c] = src[((size_t)b * srcL + l) * DM + c];
}

// ---------------------------------------------------------------------------
// softmax over contiguous n (= Lv*P <= 12) logits per (b,l,h), in place
// ---------------------------------------------------------------------------
__global__ void k_softmax(float* __restrict__ a, int Lcur, int n) {
    size_t i = (size_t)blockIdx.x * 256 + threadIdx.x;
    size_t tot = (size_t)NB * Lcur * NH;
    if (i >= tot) return;
    float* p = a + i * n;
    float mx = p[0];
    for (int j = 1; j < n; ++j) mx = fmaxf(mx, p[j]);
    float s = 0.f;
    for (int j = 0; j < n; ++j) { float e = expf(p[j] - mx); p[j] = e; s += e; }
    float inv = 1.f / s;
    for (int j = 0; j < n; ++j) p[j] *= inv;
}

// ---------------------------------------------------------------------------
// MS-deformable sampling. One wave per (b,l,h); lanes = 32 head channels.
// value: (B,Lcur,256) f32 token-major; samp out: (B,Lcur,256) f16 (feeds Wout GEMM)
// ---------------------------------------------------------------------------
__global__ __launch_bounds__(256)
void k_msda_sample(const float* __restrict__ val, const float* __restrict__ offs,
                   const float* __restrict__ attn, _Float16* __restrict__ samp,
                   int Lcur, int Lv) {
    const int dims[3]   = {24, 48, 96};
    const int starts[3] = {0, L5, L5 + L4};
    int lane = threadIdx.x & 31;
    int wv   = threadIdx.x >> 5;
    int idx = blockIdx.x * 8 + wv;
    if (idx >= NB * Lcur * NH) return;
    int h = idx % NH;
    int t = idx / NH;
    int l = t % Lcur;
    int b = t / Lcur;
    int lt = (l < L5) ? 0 : ((l < L5 + L4) ? 1 : 2);
    int Wg = dims[lt];
    int li = l - starts[lt];
    float rx = ((li % Wg) + 0.5f) / (float)Wg;
    float ry = ((li / Wg) + 0.5f) / (float)Wg;

    const float* op = offs + ((size_t)(b * Lcur + l) * NH + h) * (size_t)(Lv * NP * 2);
    const float* ap = attn + ((size_t)(b * Lcur + l) * NH + h) * (size_t)(Lv * NP);
    int c = h * 32 + lane;
    float acc = 0.f;
    for (int lv = 0; lv < Lv; ++lv) {
        int Wl = dims[lv];
        int stv = starts[lv];
        float invW = 1.0f / (float)Wl;
        const float* vb = val + (size_t)b * Lcur * DM;
        for (int p = 0; p < NP; ++p) {
            float ox = op[(lv * NP + p) * 2 + 0];
            float oy = op[(lv * NP + p) * 2 + 1];
            float w  = ap[lv * NP + p];
            float sx = (rx + ox * invW) * (float)Wl - 0.5f;
            float sy = (ry + oy * invW) * (float)Wl - 0.5f;
            float fx = floorf(sx), fy = floorf(sy);
            int x0 = (int)fx, y0 = (int)fy;
            float wx = sx - fx, wy = sy - fy;
            float v00 = 0.f, v10 = 0.f, v01 = 0.f, v11 = 0.f;
            if (y0 >= 0 && y0 < Wl) {
                if (x0 >= 0 && x0 < Wl)         v00 = vb[((size_t)(stv + y0 * Wl + x0)) * DM + c];
                if (x0 + 1 >= 0 && x0 + 1 < Wl) v10 = vb[((size_t)(stv + y0 * Wl + x0 + 1)) * DM + c];
            }
            if (y0 + 1 >= 0 && y0 + 1 < Wl) {
                if (x0 >= 0 && x0 < Wl)         v01 = vb[((size_t)(stv + (y0 + 1) * Wl + x0)) * DM + c];
                if (x0 + 1 >= 0 && x0 + 1 < Wl) v11 = vb[((size_t)(stv + (y0 + 1) * Wl + x0 + 1)) * DM + c];
            }
            acc += w * ((1.f - wx) * (1.f - wy) * v00 + wx * (1.f - wy) * v10 +
                        (1.f - wx) * wy * v01 + wx * wy * v11);
        }
    }
    samp[(size_t)(b * Lcur + l) * DM + c] = (_Float16)acc;
}

// ---------------------------------------------------------------------------
// LayerNorm of (x + y): one wave per token row; f32 out to dst[b*dstL + l],
// optional f16 copy at compact rows (feeds the next GEMM).
// ---------------------------------------------------------------------------
__global__ __launch_bounds__(256)
void k_ln_res(const float* __restrict__ x, const float* __restrict__ y,
              const float* __restrict__ g, const float* __restrict__ beta,
              float* __restrict__ out, _Float16* __restrict__ outh,
              int Lcur, int dstL) {
    int lane = threadIdx.x & 31;
    int wv   = threadIdx.x >> 5;
    int row = blockIdx.x * 8 + wv;
    if (row >= NB * Lcur) return;
    int b = row / Lcur, l = row % Lcur;
    const float* xr = x + (size_t)row * DM;
    const float* yr = y + (size_t)row * DM;
    float v[8];
    float s = 0.f;
    #pragma unroll
    for (int i = 0; i < 8; ++i) { v[i] = xr[lane + i * 32] + yr[lane + i * 32]; s += v[i]; }
    s = waveReduceSum(s);
    float mean = s * (1.0f / DM);
    float ss = 0.f;
    #pragma unroll
    for (int i = 0; i < 8; ++i) { float d = v[i] - mean; ss += d * d; }
    ss = waveReduceSum(ss);
    float rstd = rsqrtf(ss * (1.0f / DM) + 1e-5f);
    float* o = out + ((size_t)b * dstL + l) * DM;
    _Float16* oh = outh ? outh + (size_t)row * DM : nullptr;
    #pragma unroll
    for (int i = 0; i < 8; ++i) {
        int c = lane + i * 32;
        float r = (v[i] - mean) * rstd * g[c] + beta[c];
        o[c] = r;
        if (oh) oh[c] = (_Float16)r;
    }
}

// ---------------------------------------------------------------------------
// WMMA GEMM: C(M,N) = A16(M,K) * W(K,N) [BT=(N,K) f16] + bias, opt ReLU.
// One wave computes NT consecutive 16x16 N-tiles for one M-tile, so the A
// fragment is reused across NT WMMAs. 8 waves/block (8 M-tiles).
// K%32==0, M%16==0, (N/16)%NT==0.
// Fragment layouts used here (A/B in contiguous k order; C as in CDNA5 ISA §7.12.2):
//   A (16x32 f16): lane m=lane&15, kgrp=lane>>4; elem j -> K = kgrp*16 + j
//   B (32x16 f16): lane n=lane&15, kgrp=lane>>4; elem j -> K = kgrp*16 + j (contig 32B)
//   C (16x16 f32): lane n=lane&15; vgpr r -> M = r + 8*(lane>>4)
// ---------------------------------------------------------------------------
template <int NT>
__global__ __launch_bounds__(256)
void k_gemm_wmma(const _Float16* __restrict__ A, const _Float16* __restrict__ BT,
                 const float* __restrict__ bias, float* __restrict__ Cf,
                 _Float16* __restrict__ Ch, int M, int N, int K, int relu) {
    int lane = threadIdx.x & 31;
    int wv   = threadIdx.x >> 5;
    int nbase = blockIdx.x * (16 * NT);
    int tileM = blockIdx.y * 8 + wv;
    if (tileM * 16 >= M) return;
    int rowm = lane & 15;
    int kgrp = lane >> 4;
    const _Float16* arow = A + (size_t)(tileM * 16 + rowm) * K;
    v8f acc[NT];
    #pragma unroll
    for (int nt = 0; nt < NT; ++nt) acc[nt] = (v8f){};

    for (int ks = 0; ks < K; ks += 32) {
        const _Float16* ap = arow + ks + kgrp * 16;
        h8 a0 = *(const h8*)(ap);
        h8 a1 = *(const h8*)(ap + 8);
        v16h af;
        #pragma unroll
        for (int j = 0; j < 8; ++j) { af[j] = a0[j]; af[j + 8] = a1[j]; }
        #pragma unroll
        for (int nt = 0; nt < NT; ++nt) {
            const _Float16* bp = BT + (size_t)(nbase + nt * 16 + rowm) * K + ks + kgrp * 16;
            v16h bf = *(const v16h*)bp;
            acc[nt] = __builtin_amdgcn_wmma_f32_16x16x32_f16(
                false, af, false, bf, (short)0, acc[nt], false, false);
        }
    }
    int mbase = tileM * 16 + ((lane >> 4) ? 8 : 0);
    #pragma unroll
    for (int nt = 0; nt < NT; ++nt) {
        int nc = nbase + nt * 16 + (lane & 15);
        float bv = bias ? bias[nc] : 0.f;
        #pragma unroll
        for (int r = 0; r < 8; ++r) {
            float o = acc[nt][r] + bv;
            if (relu) o = fmaxf(o, 0.f);
            if (Cf) Cf[(size_t)(mbase + r) * N + nc] = o;
            if (Ch) Ch[(size_t)(mbase + r) * N + nc] = (_Float16)o;
        }
    }
}

// ---------------------------------------------------------------------------
// Host side
// ---------------------------------------------------------------------------
static inline unsigned cdiv(size_t a, size_t b) { return (unsigned)((a + b - 1) / b); }

extern "C" void kernel_launch(void* const* d_in, const int* in_sizes, int n_in,
                              void* d_out, int out_size, void* d_ws, size_t ws_size,
                              hipStream_t stream) {
    (void)in_sizes; (void)n_in; (void)out_size; (void)ws_size;

    // ---- inputs (setup_inputs dict order) ----
    const float* SRC[3]  = {(const float*)d_in[0], (const float*)d_in[2], (const float*)d_in[4]};
    const float* POSI[3] = {(const float*)d_in[1], (const float*)d_in[3], (const float*)d_in[5]};
    const float* LVLEMB  = (const float*)d_in[6];
    const float* TRCW    = (const float*)d_in[7];
    const float* TRCB    = (const float*)d_in[8];
    auto LP = [&](int layer, int k) -> const float* {
        return (const float*)d_in[9 + 16 * layer + k];
    };

    // ---- workspace layout (byte offsets, 256B aligned) ----
    char* wsb = (char*)d_ws;
    size_t o = 0;
    auto allocB = [&](size_t bytes) { size_t r = o; o += (bytes + 255) & ~(size_t)255; return r; };
    const size_t NTOK = (size_t)NB * LT;                  // 24192 rows max
    const size_t B_POS   = allocB(NTOK * DM * 4);
    const size_t B_RES5  = allocB((size_t)NB * L5 * DM * 4);
    const size_t B_RES4  = allocB((size_t)NB * L4 * DM * 4);
    const size_t B_RES3  = allocB((size_t)NB * L3 * DM * 4);
    const size_t B_RES4G = allocB((size_t)NB * L4 * DM * 4);
    const size_t B_X0    = allocB((size_t)NB * L5 * DM * 4);
    const size_t B_X1    = allocB((size_t)NB * 2880 * DM * 4);
    const size_t B_X2    = allocB(NTOK * DM * 4);
    const size_t B_OFFS  = allocB(NTOK * 192 * 4);
    const size_t B_ATTN  = allocB(NTOK * 96 * 4);
    const size_t B_AOUT  = allocB(NTOK * DM * 4);
    const size_t B_H1    = allocB(NTOK * DM * 4);
    const size_t B_H116  = allocB(NTOK * DM * 2);
    const size_t B_TRCT  = allocB((size_t)NB * DM * L4 * 4);
    // alias block: [X16 | Q16 | SAMP16 | VAL] -- all dead before the FFN hidden
    // (f16, NTOK x 1024 = 49.5MB) is written starting at B_X16.
    const size_t B_X16   = allocB(NTOK * DM * 2);
    const size_t B_Q16   = allocB(NTOK * DM * 2);
    const size_t B_SAMP16= allocB(NTOK * DM * 2);
    const size_t B_VAL   = allocB(NTOK * DM * 4);
    const size_t B_FFN16 = B_X16;   // needs NTOK*1024*2 <= (3*NTOK*DM*2 + NTOK*DM*4)
    const size_t B_WPACK = allocB((size_t)3 * 800 * 1024 * 2);  // packed weights, generous

    float*    POSF  = (float*)(wsb + B_POS);
    float*    RES5  = (float*)(wsb + B_RES5);
    float*    RES4  = (float*)(wsb + B_RES4);
    float*    RES3  = (float*)(wsb + B_RES3);
    float*    RES4G = (float*)(wsb + B_RES4G);
    float*    OFFS  = (float*)(wsb + B_OFFS);
    float*    ATTN  = (float*)(wsb + B_ATTN);
    float*    AOUT  = (float*)(wsb + B_AOUT);
    float*    H1    = (float*)(wsb + B_H1);
    float*    VAL   = (float*)(wsb + B_VAL);
    float*    TRCT  = (float*)(wsb + B_TRCT);
    _Float16* X16   = (_Float16*)(wsb + B_X16);
    _Float16* Q16   = (_Float16*)(wsb + B_Q16);
    _Float16* SAMP16= (_Float16*)(wsb + B_SAMP16);
    _Float16* H116  = (_Float16*)(wsb + B_H116);
    _Float16* FFN16 = (_Float16*)(wsb + B_FFN16);

    _Float16* wp = (_Float16*)(wsb + B_WPACK);
    size_t wo = 0;
    _Float16 *WvT[3], *WoffT[3], *WaT[3], *WoutT[3], *W1T[3], *W2T[3];
    for (int i = 0; i < 3; ++i) {
        int Lv = i + 1;
        WvT[i]   = wp + wo; wo += (size_t)256 * 256;
        WoffT[i] = wp + wo; wo += (size_t)64 * Lv * 256;
        WaT[i]   = wp + wo; wo += (size_t)32 * Lv * 256;
        WoutT[i] = wp + wo; wo += (size_t)256 * 256;
        W1T[i]   = wp + wo; wo += (size_t)1024 * 256;
        W2T[i]   = wp + wo; wo += (size_t)256 * 1024;
    }

    dim3 B256(256);
    auto pack = [&](const float* W, _Float16* dst, int K, int N) {
        k_pack_weight<<<cdiv((size_t)K * N, 256), B256, 0, stream>>>(W, dst, K, N);
    };
    auto gemm = [&](const _Float16* A, const _Float16* BT, const float* bias,
                    float* Cf, _Float16* Ch, int M, int N, int K, int relu) {
        int ntiles = N / 16;
        if (ntiles % 4 == 0) {
            dim3 g((unsigned)(ntiles / 4), (unsigned)((M / 16 + 7) / 8));
            k_gemm_wmma<4><<<g, B256, 0, stream>>>(A, BT, bias, Cf, Ch, M, N, K, relu);
        } else {
            dim3 g((unsigned)(ntiles / 2), (unsigned)((M / 16 + 7) / 8));
            k_gemm_wmma<2><<<g, B256, 0, stream>>>(A, BT, bias, Cf, Ch, M, N, K, relu);
        }
    };

    // ---- 1. pack all weights ----
    for (int i = 0; i < 3; ++i) {
        int Lv = i + 1;
        pack(LP(i, 4),  WvT[i],   256, 256);
        pack(LP(i, 0),  WoffT[i], 256, 64 * Lv);
        pack(LP(i, 2),  WaT[i],   256, 32 * Lv);
        pack(LP(i, 6),  WoutT[i], 256, 256);
        pack(LP(i, 10), W1T[i],   256, 1024);
        pack(LP(i, 12), W2T[i],   1024, 256);
    }

    // ---- 2. flatten srcs, build full pos (+level embeds) ----
    const int Ls[3] = {L5, L4, L3};
    const int starts3[3] = {0, L5, L5 + L4};
    float* FRES[3] = {RES5, RES4, RES3};
    for (int lv = 0; lv < 3; ++lv) {
        size_t tot = (size_t)NB * Ls[lv] * DM;
        k_flatten<<<cdiv(tot, 256), B256, 0, stream>>>(SRC[lv], FRES[lv], Ls[lv]);
        k_flatten_pos<<<cdiv(tot, 256), B256, 0, stream>>>(
            POSI[lv], LVLEMB + (size_t)lv * DM, POSF, Ls[lv], starts3[lv]);
    }

    // ---- 3. TRC gating of res5 and res4 against res3 ----
    float* X0 = (float*)(wsb + B_X0);
    float* X1 = (float*)(wsb + B_X1);
    float* X2 = (float*)(wsb + B_X2);
    k_resize<<<cdiv((size_t)NB * DM * 24 * 24, 256), B256, 0, stream>>>(RES3, TRCT, 24);
    k_trc_gate<<<cdiv((size_t)NB * L5, 8), B256, 0, stream>>>(RES5, TRCT, TRCW, TRCB, X0, L5);
    k_resize<<<cdiv((size_t)NB * DM * 48 * 48, 256), B256, 0, stream>>>(RES3, TRCT, 48);
    k_trc_gate<<<cdiv((size_t)NB * L4, 8), B256, 0, stream>>>(RES4, TRCT, TRCW, TRCB, RES4G, L4);

    // ---- 4. three encoder layers ----
    const int Lcur[3] = {576, 2880, 12096};
    float* xin[3]  = {X0, X1, X2};
    float* xout[3] = {X1, X2, (float*)d_out};
    const int dstL[3] = {2880, 12096, 12096};

    for (int i = 0; i < 3; ++i) {
        int Lc = Lcur[i];
        int Lv = i + 1;
        int M  = NB * Lc;
        float* X = xin[i];

        // residual stream -> f16 staging; q16 = f16(x + pos)
        k_cvt_f16<<<cdiv((size_t)M * DM, 256), B256, 0, stream>>>(X, X16, (size_t)M * DM);
        k_add_pos<<<cdiv((size_t)M * DM, 256), B256, 0, stream>>>(X, POSF, Q16, Lc);
        // value / offsets / attention logits
        gemm(X16, WvT[i],   LP(i, 5), VAL,  nullptr, M, 256,     256, 0);
        gemm(Q16, WoffT[i], LP(i, 1), OFFS, nullptr, M, 64 * Lv, 256, 0);
        gemm(Q16, WaT[i],   LP(i, 3), ATTN, nullptr, M, 32 * Lv, 256, 0);
        k_softmax<<<cdiv((size_t)M * NH, 256), B256, 0, stream>>>(ATTN, Lc, Lv * NP);
        // deformable sampling -> f16 (B,Lc,256)
        k_msda_sample<<<cdiv((size_t)M * NH, 8), B256, 0, stream>>>(VAL, OFFS, ATTN, SAMP16, Lc, Lv);
        // output projection + residual LN (f32 h1 + f16 copy for the FFN GEMM)
        gemm(SAMP16, WoutT[i], LP(i, 7), AOUT, nullptr, M, 256, 256, 0);
        k_ln_res<<<cdiv((size_t)M, 8), B256, 0, stream>>>(X, AOUT, LP(i, 8), LP(i, 9),
                                                          H1, H116, Lc, Lc);
        // FFN: hidden produced f16-only (aliases dead X16/Q16/SAMP16/VAL block)
        gemm(H116, W1T[i], LP(i, 11), nullptr, FFN16, M, 1024, 256, 1);
        gemm(FFN16, W2T[i], LP(i, 13), AOUT, nullptr, M, 256, 1024, 0);
        k_ln_res<<<cdiv((size_t)M, 8), B256, 0, stream>>>(H1, AOUT, LP(i, 14), LP(i, 15),
                                                          xout[i], nullptr, Lc, dstL[i]);

        // concat next resolution level after the layer output
        if (i == 0) {
            k_copy_rows<<<cdiv((size_t)NB * L4 * DM, 256), B256, 0, stream>>>(
                RES4G, X1, L4, L4, 2880, 576);
        } else if (i == 1) {
            k_copy_rows<<<cdiv((size_t)NB * L3 * DM, 256), B256, 0, stream>>>(
                RES3, X2, L3, L3, 12096, 2880);
        }
    }
}
